// AttentionScores_2396591751173
// MI455X (gfx1250) — matrix operand order repaired and checked by reading, not compiled, not run on hardware
//
#include <hip/hip_runtime.h>
#include <stdint.h>

// ---------------- CDNA5 WMMA / TDM types ----------------
typedef __attribute__((ext_vector_type(16))) __bf16       v16bf;
typedef __attribute__((ext_vector_type(8)))  float        v8f;
typedef __attribute__((ext_vector_type(4)))  unsigned int v4u;
typedef __attribute__((ext_vector_type(4)))  unsigned int u32x4;
typedef __attribute__((ext_vector_type(8)))  int          i32x8;
typedef __attribute__((ext_vector_type(4)))  int          i32x4;
typedef unsigned short u16;

#define SEQ   2048
#define BATCH 4
#define DMODEL 1024
#define NHEADS 16
#define HDIM  64
#define MTOT  (BATCH * SEQ)   // 8192
#define KC    128             // K-chunk staged to LDS by the TDM in GEMM

// fp32 -> bf16 round-to-nearest-even
static __device__ __forceinline__ u16 f2bf(float f) {
  union { float f; unsigned int u; } c; c.f = f;
  unsigned int r = c.u + 0x7FFFu + ((c.u >> 16) & 1u);
  return (u16)(r >> 16);
}

static __device__ __forceinline__ v8f wmma_bf16(v16bf a, v16bf b, v8f c) {
  // (neg_a, A, neg_b, B, c_mod, C, reuse_a, reuse_b)
  return __builtin_amdgcn_wmma_f32_16x16x32_bf16(false, a, false, b, (short)0, c,
                                                 false, false);
}

// A fragment: 16x32 bf16, row-major source, leading dim ld (elements).
// lane<16: row=lane,    K = {0..7, 16..23}
// lane>=16: row=lane-16, K = {8..15, 24..31}
static __device__ __forceinline__ v16bf load_frag_a(const u16* base, int ld, int lane) {
  int m  = lane & 15;
  int hi = lane >> 4;
  v16bf r;
  v4u* rp = (v4u*)&r;
  rp[0] = *(const v4u*)(base + (size_t)m * ld + hi * 8);
  rp[1] = *(const v4u*)(base + (size_t)m * ld + 16 + hi * 8);
  return r;
}

// B fragment: 32x16 bf16. Source holds column n as a contiguous row (i.e. B^T
// row-major with leading dim ld). lane<16: n=lane, K={0..7,16..23}; lane>=16: n=lane-16, K={8..15,24..31}
static __device__ __forceinline__ v16bf load_frag_b(const u16* base, int ld, int lane) {
  int n  = lane & 15;
  int hi = lane >> 4;
  const u16* p = base + (size_t)n * ld + hi * 8;
  v16bf r;
  v4u* rp = (v4u*)&r;
  rp[0] = *(const v4u*)p;
  rp[1] = *(const v4u*)(p + 16);
  return r;
}

// ---------------- Tensor Data Mover: 2-D tile Global -> LDS ----------------
// Builds a D# per CDNA5 ISA ch.8 and issues TENSOR_LOAD_TO_LDS.
// tile_w elements (data_size=2B) per row, tile_h rows, row stride = stride0 elems.
static __device__ __forceinline__ void tdm_load_2d(unsigned lds_off, const void* gptr,
                                                   unsigned tile_w, unsigned tile_h,
                                                   unsigned stride0) {
  unsigned long long ga = (unsigned long long)gptr;
  u32x4 g0;
  g0[0] = 0x1u;                                   // count=1, user mode
  g0[1] = lds_off;                                // lds_addr (bytes)
  g0[2] = (unsigned)(ga & 0xFFFFFFFFu);           // global_addr[31:0]
  g0[3] = (unsigned)((ga >> 32) & 0x01FFFFFFu)    // global_addr[56:32]
          | (2u << 30);                           // type=2 ("image")
  i32x8 g1;
  g1[0] = (int)(1u << 16);                        // data_size=1 (2 bytes), mask=0
  g1[1] = (int)((tile_w & 0xFFFFu) << 16);        // tensor_dim0[15:0] = tile_w
  g1[2] = (int)(((tile_w >> 16) & 0xFFFFu) | ((tile_h & 0xFFFFu) << 16)); // dim0 hi | dim1 lo
  g1[3] = (int)(((tile_h >> 16) & 0xFFFFu) | ((tile_w & 0xFFFFu) << 16)); // dim1 hi | tile_dim0
  g1[4] = (int)(tile_h & 0xFFFFu);                // tile_dim1; tile_dim2 = 0
  g1[5] = (int)stride0;                           // tensor_dim0_stride[31:0]
  g1[6] = 0;                                      // stride0[47:32] | stride1[15:0]
  g1[7] = 0;                                      // stride1[47:16]
  i32x4 z4 = {0, 0, 0, 0};                        // groups 2/3: 2-D tensor
#if __clang_major__ >= 23
  i32x8 z8 = {0, 0, 0, 0, 0, 0, 0, 0};
  __builtin_amdgcn_tensor_load_to_lds(g0, g1, z4, z4, z8, 0);
#else
  __builtin_amdgcn_tensor_load_to_lds(g0, g1, z4, z4, 0);
#endif
}

// ---------------- stage 0: conversions ----------------
__global__ void k_f32_to_bf16(const float* __restrict__ in, u16* __restrict__ out, int n) {
  int i = blockIdx.x * blockDim.x + threadIdx.x;
  int stride = gridDim.x * blockDim.x;
  for (; i < n; i += stride) out[i] = f2bf(in[i]);
}

// wt[n*dim + k] = w[k*dim + n]   (store W^T as bf16)
__global__ void k_transpose_bf16(const float* __restrict__ w, u16* __restrict__ wt, int dim) {
  int i = blockIdx.x * blockDim.x + threadIdx.x;
  int stride = gridDim.x * blockDim.x;
  int tot = dim * dim;
  for (; i < tot; i += stride) {
    int n = i / dim, k = i - n * dim;
    wt[i] = f2bf(w[(size_t)k * dim + n]);
  }
}

// ---------------- stage 1/3: WMMA GEMM  C[M,N] = A[M,K] * B^T-rows[N,K] ----------------
// MODE 0: store bf16 row-major [M,N]
// MODE 1: store bf16 transposed [N, ldT]  (for V^T)
// MODE 2: store fp32 row-major + bias
// block = 128 threads (4 waves); block tile 64 rows x 64 cols; wave tile 16x64.
// The 64x128 B panel (shared by all 4 waves) is staged to LDS by the TDM.
template <int MODE>
__global__ __launch_bounds__(128) void k_gemm_wmma(const u16* __restrict__ A,
                                                   const u16* __restrict__ Bt,
                                                   void* __restrict__ Cv,
                                                   const float* __restrict__ bias,
                                                   int K, int N, int ldT) {
  __shared__ u16 ldsB[64 * KC];  // 16 KB: 64 B^T rows x 128 K elements

  const int lane = threadIdx.x & 31;
  const int wave = threadIdx.x >> 5;
  const int row0 = blockIdx.y * 64 + wave * 16;
  const int col0 = blockIdx.x * 64;

  v8f acc[4];
#pragma unroll
  for (int j = 0; j < 4; ++j)
#pragma unroll
    for (int r = 0; r < 8; ++r) acc[j][r] = 0.0f;

  const u16* aRow = A + (size_t)row0 * K;
  const unsigned ldsOff = (unsigned)(unsigned long long)(void*)&ldsB[0];

  for (int kc = 0; kc < K; kc += KC) {
    __syncthreads();  // previous chunk fully consumed before overwrite
    if (wave == 0) {
      tdm_load_2d(ldsOff, Bt + (size_t)col0 * K + kc, KC, 64, (unsigned)K);
      __builtin_amdgcn_s_wait_tensorcnt(0);
    }
    __syncthreads();  // publish LDS tile to all waves

#pragma unroll
    for (int kk = 0; kk < KC; kk += 32) {
      v16bf af = load_frag_a(aRow + kc + kk, K, lane);
      if (kc + kk + 32 < K)
        __builtin_prefetch(aRow + (size_t)(lane & 15) * K + kc + kk + 32, 0, 1);
#pragma unroll
      for (int j = 0; j < 4; ++j) {
        v16bf bf = load_frag_b(ldsB + (size_t)(j * 16) * KC + kk, KC, lane);
        acc[j] = wmma_bf16(af, bf, acc[j]);
      }
    }
  }

  const int n   = lane & 15;
  const int mhi = (lane >> 4) * 8;
#pragma unroll
  for (int j = 0; j < 4; ++j) {
#pragma unroll
    for (int r = 0; r < 8; ++r) {
      int m = row0 + mhi + r;
      int c = col0 + j * 16 + n;
      if (MODE == 0) {
        ((u16*)Cv)[(size_t)m * N + c] = f2bf(acc[j][r]);
      } else if (MODE == 1) {
        ((u16*)Cv)[(size_t)c * ldT + m] = f2bf(acc[j][r]);
      } else {
        ((float*)Cv)[(size_t)m * N + c] = acc[j][r] + bias[c];
      }
    }
  }
}

// ---------------- stage 2: causal flash attention ----------------
// grid = (SEQ/16, NHEADS, BATCH), block = 32 (one wave / query tile).
// Q,K: bf16 [MTOT, DMODEL] row-major. Vt: bf16 [DMODEL, MTOT] (V transposed).
// Ctx: bf16 [MTOT, DMODEL].
__global__ __launch_bounds__(32) void k_attn(const u16* __restrict__ Q,
                                             const u16* __restrict__ Kx,
                                             const u16* __restrict__ Vt,
                                             u16* __restrict__ Ctx) {
  __shared__ u16 ldsP[16 * 32];  // P tile, 16 queries x 32 keys, row-major

  const int lane = threadIdx.x;
  const int qt = blockIdx.x, h = blockIdx.y, b = blockIdx.z;

  const u16* Qb = Q  + (size_t)b * SEQ * DMODEL + h * HDIM;
  const u16* Kb = Kx + (size_t)b * SEQ * DMODEL + h * HDIM;
  const u16* Vb = Vt + (size_t)h * HDIM * MTOT + (size_t)b * SEQ;  // row stride MTOT

  // Q fragments for this query tile (d = 0..31 and 32..63), loaded once.
  const u16* qRow = Qb + (size_t)(qt * 16) * DMODEL;
  v16bf aq0 = load_frag_a(qRow, DMODEL, lane);
  v16bf aq1 = load_frag_a(qRow + 32, DMODEL, lane);

  v8f o[4];
  float rm[8], rs[8];
#pragma unroll
  for (int j = 0; j < 4; ++j)
#pragma unroll
    for (int r = 0; r < 8; ++r) o[j][r] = 0.0f;
#pragma unroll
  for (int r = 0; r < 8; ++r) { rm[r] = -__builtin_inff(); rs[r] = 0.0f; }

  v8f zf;
#pragma unroll
  for (int r = 0; r < 8; ++r) zf[r] = 0.0f;

  const float scale = 0.125f;  // 1/sqrt(64)
  const int mhi = (lane >> 4) * 8;
  const int nl  = lane & 15;
  const int nkb = (16 * qt + 47) >> 5;  // # of 32-key blocks (causal)

  for (int kb = 0; kb < nkb; ++kb) {
    const int kbase = kb * 32;

    // ---- scores for two 16-key tiles, K-dim = HDIM = 64 (2 wmma each) ----
    v8f s0 = wmma_bf16(aq0, load_frag_b(Kb + (size_t)kbase * DMODEL, DMODEL, lane), zf);
    s0 = wmma_bf16(aq1, load_frag_b(Kb + (size_t)kbase * DMODEL + 32, DMODEL, lane), s0);
    v8f s1 = wmma_bf16(aq0, load_frag_b(Kb + (size_t)(kbase + 16) * DMODEL, DMODEL, lane), zf);
    s1 = wmma_bf16(aq1, load_frag_b(Kb + (size_t)(kbase + 16) * DMODEL + 32, DMODEL, lane), s1);

    // ---- online softmax (fp32), rows live per half-wave ----
#pragma unroll
    for (int r = 0; r < 8; ++r) {
      const int mg = qt * 16 + mhi + r;                 // global query index
      float a0 = s0[r] * scale;
      float a1 = s1[r] * scale;
      if (kbase + nl > mg)      a0 = -__builtin_inff(); // causal mask
      if (kbase + 16 + nl > mg) a1 = -__builtin_inff();

      float mloc = fmaxf(a0, a1);
      for (int msk = 1; msk < 16; msk <<= 1)
        mloc = fmaxf(mloc, __shfl_xor(mloc, msk, 32));
      float newm  = fmaxf(rm[r], mloc);
      float alpha = __expf(rm[r] - newm);
      rm[r] = newm;

      float p0 = __expf(a0 - newm);
      float p1 = __expf(a1 - newm);
      float lsum = p0 + p1;
      for (int msk = 1; msk < 16; msk <<= 1)
        lsum += __shfl_xor(lsum, msk, 32);
      rs[r] = rs[r] * alpha + lsum;

#pragma unroll
      for (int j = 0; j < 4; ++j) o[j][r] *= alpha;

      // stage P tile to LDS (bf16, 16x32 row-major) for C->A layout change
      ldsP[(mhi + r) * 32 + nl]      = f2bf(p0);
      ldsP[(mhi + r) * 32 + 16 + nl] = f2bf(p1);
    }

    asm volatile("s_wait_dscnt 0" ::: "memory");

    // ---- ctx += P(16x32) * V(32x64): A from LDS, B rows from V^T ----
    v16bf ap = load_frag_a(ldsP, 32, lane);
#pragma unroll
    for (int j = 0; j < 4; ++j) {
      v16bf bv = load_frag_b(Vb + (size_t)(j * 16) * MTOT + kbase, MTOT, lane);
      o[j] = wmma_bf16(ap, bv, o[j]);
    }
    asm volatile("" ::: "memory");  // keep LDS reuse ordered across iterations
  }

  // ---- normalize and store ctx (bf16) ----
#pragma unroll
  for (int r = 0; r < 8; ++r) {
    float inv = 1.0f / rs[r];
#pragma unroll
    for (int j = 0; j < 4; ++j) {
      float v = o[j][r] * inv;
      size_t m = (size_t)b * SEQ + qt * 16 + mhi + r;
      Ctx[m * DMODEL + h * HDIM + j * 16 + nl] = f2bf(v);
    }
  }
}

// ---------------- host-side launcher ----------------
extern "C" void kernel_launch(void* const* d_in, const int* in_sizes, int n_in,
                              void* d_out, int out_size, void* d_ws, size_t ws_size,
                              hipStream_t stream) {
  const float* X  = (const float*)d_in[0];
  const float* Wq = (const float*)d_in[1];
  const float* Wk = (const float*)d_in[2];
  const float* Wv = (const float*)d_in[3];
  const float* Wo = (const float*)d_in[4];
  const float* bo = (const float*)d_in[5];
  float* out = (float*)d_out;

  char* ws = (char*)d_ws;
  const size_t MB = 1u << 20;
  u16* xbf = (u16*)(ws + 0);          // 16 MB  [MTOT, DMODEL]
  u16* wqT = (u16*)(ws + 16 * MB);    //  2 MB  [DMODEL, DMODEL] transposed
  u16* wkT = (u16*)(ws + 18 * MB);
  u16* wvT = (u16*)(ws + 20 * MB);
  u16* woT = (u16*)(ws + 22 * MB);
  u16* qbf = (u16*)(ws + 24 * MB);    // 16 MB
  u16* kbf = (u16*)(ws + 40 * MB);    // 16 MB
  u16* vtb = (u16*)(ws + 56 * MB);    // 16 MB  [DMODEL, MTOT] (V^T)
  u16* ctx = xbf;                     // reuse X buffer after projections

  const int nX = MTOT * DMODEL;       // 8388608

  // stage 0: conversions
  k_f32_to_bf16<<<4096, 256, 0, stream>>>(X, xbf, nX);
  k_transpose_bf16<<<4096, 256, 0, stream>>>(Wq, wqT, DMODEL);
  k_transpose_bf16<<<4096, 256, 0, stream>>>(Wk, wkT, DMODEL);
  k_transpose_bf16<<<4096, 256, 0, stream>>>(Wv, wvT, DMODEL);
  k_transpose_bf16<<<4096, 256, 0, stream>>>(Wo, woT, DMODEL);

  // stage 1: Q/K/V projections
  dim3 ggrid(DMODEL / 64, MTOT / 64);  // (16, 128)
  k_gemm_wmma<0><<<ggrid, 128, 0, stream>>>(xbf, wqT, (void*)qbf, nullptr, DMODEL, DMODEL, 0);
  k_gemm_wmma<0><<<ggrid, 128, 0, stream>>>(xbf, wkT, (void*)kbf, nullptr, DMODEL, DMODEL, 0);
  k_gemm_wmma<1><<<ggrid, 128, 0, stream>>>(xbf, wvT, (void*)vtb, nullptr, DMODEL, DMODEL, MTOT);

  // stage 2: causal flash attention (one wave per 16-query tile)
  dim3 agrid(SEQ / 16, NHEADS, BATCH);  // (128, 16, 4)
  k_attn<<<agrid, 32, 0, stream>>>(qbf, kbf, vtb, ctx);

  // stage 3: output projection + bias (fp32 out)
  k_gemm_wmma<2><<<ggrid, 128, 0, stream>>>(ctx, woT, (void*)out, bo, DMODEL, DMODEL, 0);
}
